// EquivariantDecoder_28089086116408
// MI455X (gfx1250) — matrix operand order repaired and checked by reading, not compiled, not run on hardware
//
#include <hip/hip_runtime.h>

typedef __attribute__((ext_vector_type(16))) __bf16          v16bf;
typedef __attribute__((ext_vector_type(8)))  float           v8f;
typedef __attribute__((ext_vector_type(4)))  float           v4f;
typedef __attribute__((ext_vector_type(8)))  unsigned short  u16x8;
typedef __attribute__((ext_vector_type(16))) unsigned short  u16x16;

#define TILE_B 16
#define BLOCK  256
#define NWAVE  8

// ---------------- geometry tables ----------------
// layer-1 input (IN_IRREPS) LDS layout: per (l,m) a 16 x mul matrix, row stride = mul+8
__device__ const int g_K1[7]    = {256,128,128,64,64,64,64};
__device__ const int g_KS1[7]   = {8,4,4,2,2,2,2};            // K/32
__device__ const int g_RS1[7]   = {264,136,136,72,72,72,72};
__device__ const int g_AOFF1[7] = {0,4224,10752,21632,29696,40064,52736};   // halfs, total 67712

// hidden (HID_IRREPS) LDS layout
__device__ const int g_KH[7]    = {64,64,64,32,32,32,32};
__device__ const int g_KSH[7]   = {2,2,2,1,1,1,1};
__device__ const int g_RSH[7]   = {72,72,72,40,40,40,40};
__device__ const int g_HOFF[7]  = {0,1152,4608,10368,14848,20608,27648};    // halfs, total 35968

// packed-weight (bf16, [o][k], scale folded) per-l offsets in halfs
__device__ const int g_W1OFF[7]  = {0,81920,90112,98304,100352,102400,104448};  // base 0,      106496 halfs
__device__ const int g_W23OFF[7] = {0,20480,24576,28672,29696,30720,31744};     // base 106496 / 139264
__device__ const int g_GOFF[7]   = {0,0,64,128,160,192,224};                    // gate channel offset per l

// phase-B task map: 16 (l, ntile) tasks for the l>0 GEMMs
__device__ const int g_TL[16] = {1,1,1,1,2,2,2,2,3,3,4,4,5,5,6,6};
__device__ const int g_TN[16] = {0,1,2,3,0,1,2,3,0,1,0,1,0,1,0,1};

// output segment offsets per l; layer-4 replicated weights [l][16][K] at base 172032
__device__ const int g_POFF[8]   = {0,1,4,9,16,25,36,49};
__device__ const int g_W4POFF[7] = {0,1024,2048,3072,3584,4096,4608};           // 5120 halfs total

// ---------------- helpers ----------------
__device__ __forceinline__ unsigned short f2bf(float f) {
  unsigned u = __float_as_uint(f);
  u += 0x7FFFu + ((u >> 16) & 1u);          // round-to-nearest-even
  return (unsigned short)(u >> 16);
}
__device__ __forceinline__ float fast_sigmoid(float z) {
  return __builtin_amdgcn_rcpf(1.0f + __expf(-z));   // v_exp + v_add + v_rcp
}
__device__ __forceinline__ v16bf combine16(u16x8 lo, u16x8 hi) {
  u16x16 u;
#pragma unroll
  for (int i = 0; i < 8; ++i) { u[i] = lo[i]; u[i + 8] = hi[i]; }
  return __builtin_bit_cast(v16bf, u);
}
// B fragment (32x16 bf16): col n=lane&15; lane<16 holds K {0..7,16..23}, lane>=16 K {8..15,24..31}.
__device__ __forceinline__ v16bf load_bfrag(const unsigned short* __restrict__ w,
                                            int K, int ch, int ks, int lane) {
  const unsigned short* p = w + ch * K + ks * 32 + ((lane & 16) ? 8 : 0);
  u16x8 lo = *(const u16x8*)p;
  u16x8 hi = *(const u16x8*)(p + 16);
  return combine16(lo, hi);
}
// A fragment (16x32 bf16): row = lane&15; lane<16 holds K {0..7,16..23}, lane>=16 K {8..15,24..31}.
__device__ __forceinline__ v16bf load_afrag(const unsigned short* a, int m, int RS,
                                            int ks, int lane) {
  const unsigned short* p = a + m * (RS << 4) + (lane & 15) * RS + ks * 32 + ((lane & 16) ? 8 : 0);
  u16x8 lo = *(const u16x8*)p;
  u16x8 hi = *(const u16x8*)(p + 16);
  return combine16(lo, hi);
}
__device__ __forceinline__ v8f wmma_bf16(v16bf a, v16bf b, v8f c) {
  return __builtin_amdgcn_wmma_f32_16x16x32_bf16(false, a, false, b, (short)0, c, false, false);
}

// ---------------- one e3linear + gate layer (layers 1..3) ----------------
__device__ __forceinline__ void run_layer(unsigned short* inBuf, unsigned short* outBuf,
                                          float* gates,
                                          const int* Kin, const int* KSin,
                                          const int* RSin, const int* AOFFin,
                                          const unsigned short* __restrict__ wtL,
                                          const int* WOFFL,
                                          int wave, int lane) {
  const int laneN = lane & 15;
  const int bb    = (lane & 16) ? 8 : 0;

  // ---- phase A: l=0 block -> 64 silu scalars + 256 sigmoid gates (N=320 -> 20 n-tiles)
  {
    const int K = Kin[0], nks = KSin[0], RS = RSin[0];
    const unsigned short* w  = wtL + WOFFL[0];
    const unsigned short* aB = inBuf + AOFFin[0];
    for (int t = wave; t < 20; t += NWAVE) {
      v8f acc;
#pragma unroll
      for (int r = 0; r < 8; ++r) acc[r] = 0.0f;
      const int ch = t * 16 + laneN;
      for (int ks = 0; ks < nks; ++ks) {
        v16bf bf = load_bfrag(w, K, ch, ks, lane);
        v16bf af = load_afrag(aB, 0, RS, ks, lane);
        acc = wmma_bf16(af, bf, acc);
      }
      if (t < 4) {                        // wave-uniform: whole n-tile is silu scalars
#pragma unroll
        for (int r = 0; r < 8; ++r) {
          float z = acc[r];
          outBuf[(bb + r) * 72 + ch] = f2bf(z * fast_sigmoid(z));
        }
      } else {                            // wave-uniform: whole n-tile is gates
#pragma unroll
        for (int r = 0; r < 8; ++r)
          gates[(ch - 64) * 16 + bb + r] = fast_sigmoid(acc[r]);
      }
    }
  }
  __syncthreads();

  // ---- phase B: l>0 GEMMs, gated, written to hidden-layout outBuf
  for (int t = wave; t < 16; t += NWAVE) {
    const int l = g_TL[t], nt = g_TN[t];
    const int nm = 2 * l + 1;
    const int K = Kin[l], nks = KSin[l], RS = RSin[l];
    const unsigned short* w  = wtL + WOFFL[l];
    const unsigned short* aB = inBuf + AOFFin[l];
    const int ch = nt * 16 + laneN;

    v8f acc[13];
#pragma unroll
    for (int m = 0; m < 13; ++m)
#pragma unroll
      for (int r = 0; r < 8; ++r) acc[m][r] = 0.0f;

    for (int ks = 0; ks < nks; ++ks) {
      v16bf bf = load_bfrag(w, K, ch, ks, lane);
#pragma unroll
      for (int m = 0; m < 13; ++m) {
        if (m < nm) {                       // wave-uniform guard: EXEC stays all-ones
          v16bf af = load_afrag(aB, m, RS, ks, lane);
          acc[m] = wmma_bf16(af, bf, acc[m]);
        }
      }
    }

    const int gch = g_GOFF[l] + ch;
    v4f g0 = *(const v4f*)&gates[gch * 16 + bb];
    v4f g1 = *(const v4f*)&gates[gch * 16 + bb + 4];
    float gv[8] = {g0[0], g0[1], g0[2], g0[3], g1[0], g1[1], g1[2], g1[3]};
    const int RSo = g_RSH[l], mS = RSo << 4, oB = g_HOFF[l];
#pragma unroll
    for (int m = 0; m < 13; ++m) {
      if (m < nm) {
#pragma unroll
        for (int r = 0; r < 8; ++r)
          outBuf[oB + m * mS + (bb + r) * RSo + ch] = f2bf(acc[m][r] * gv[r]);
      }
    }
  }
  __syncthreads();
}

// ---------------- fused network kernel: 1 workgroup = 16 batch rows ----------------
__global__ __launch_bounds__(BLOCK, 1)
void e3_fused_kernel(const float* __restrict__ vraw,
                     const unsigned short* __restrict__ wt,
                     float* __restrict__ out) {
  extern __shared__ unsigned char smem_raw[];
  unsigned short* bufA = (unsigned short*)smem_raw;   // 67712 halfs (layer-1 in / layer-3 in)
  unsigned short* bufB = bufA + 67712;                // 35968 halfs (hidden acts)
  float*          gates = (float*)(bufB + 35968);     // 256 ch x 16 b, f32

  const int tid  = threadIdx.x;
  const int lane = tid & 31;
  const int wave = tid >> 5;
  const int b0   = blockIdx.x * TILE_B;

  // ---- stage v_raw tile -> LDS [l][m][b][k] bf16.
  // Each thread owns 15 fixed column indices (tid + 256k); compute their
  // (LDS base, row stride) once, then sweep the 16 batch rows.
  {
    int sbase[15], srs[15];
#pragma unroll
    for (int k = 0; k < 15; ++k) {
      int idx = tid + (k << 8);
      int base, rs;
      if (idx < 256) {
        base = idx;                                              rs = 264;  // l=0
      } else if (idx < 640)  { int r = idx - 256;  int i = r / 3;  int m = r - i * 3;
        base = 4224  + m * 2176 + i;                             rs = 136;  // l=1
      } else if (idx < 1280) { int r = idx - 640;  int i = r / 5;  int m = r - i * 5;
        base = 10752 + m * 2176 + i;                             rs = 136;  // l=2
      } else if (idx < 1728) { int r = idx - 1280; int i = r / 7;  int m = r - i * 7;
        base = 21632 + m * 1152 + i;                             rs = 72;   // l=3
      } else if (idx < 2304) { int r = idx - 1728; int i = r / 9;  int m = r - i * 9;
        base = 29696 + m * 1152 + i;                             rs = 72;   // l=4
      } else if (idx < 3008) { int r = idx - 2304; int i = r / 11; int m = r - i * 11;
        base = 40064 + m * 1152 + i;                             rs = 72;   // l=5
      } else                 { int r = idx - 3008; int i = r / 13; int m = r - i * 13;
        base = 52736 + m * 1152 + i;                             rs = 72;   // l=6
      }
      sbase[k] = base; srs[k] = rs;
    }
    for (int b = 0; b < TILE_B; ++b) {
      const float* vr = vraw + (size_t)(b0 + b) * 3840;
#pragma unroll
      for (int k = 0; k < 15; ++k)
        bufA[sbase[k] + b * srs[k]] = f2bf(vr[tid + (k << 8)]);
    }
  }
  __syncthreads();

  run_layer(bufA, bufB, gates, g_K1, g_KS1, g_RS1, g_AOFF1, wt,          g_W1OFF,  wave, lane);
  run_layer(bufB, bufA, gates, g_KH, g_KSH, g_RSH, g_HOFF,  wt + 106496, g_W23OFF, wave, lane);
  run_layer(bufA, bufB, gates, g_KH, g_KSH, g_RSH, g_HOFF,  wt + 139264, g_W23OFF, wave, lane);

  // ---- layer 4 via WMMA: one (l,m) task per iteration, 49 tasks.
  // B columns all hold the same (replicated) w4 vector; only n==0 lanes store.
  for (int t = wave; t < 49; t += NWAVE) {
    int l = 0;
#pragma unroll
    for (int q = 1; q < 7; ++q) l += (t >= g_POFF[q]);
    const int m  = t - g_POFF[l];
    const int K  = g_KH[l], nks = g_KSH[l], RS = g_RSH[l];
    const unsigned short* w  = wt + 172032 + g_W4POFF[l];   // [16 cols][K]
    const unsigned short* aB = bufB + g_HOFF[l];
    v8f acc;
#pragma unroll
    for (int r = 0; r < 8; ++r) acc[r] = 0.0f;
    for (int ks = 0; ks < nks; ++ks) {
      v16bf bf = load_bfrag(w, K, (lane & 15), ks, lane);
      v16bf af = load_afrag(aB, m, RS, ks, lane);
      acc = wmma_bf16(af, bf, acc);
    }
    if ((lane & 15) == 0) {               // column 0 of D: batches bb..bb+7 in acc[r]
      const int bb = (lane & 16) ? 8 : 0;
#pragma unroll
      for (int r = 0; r < 8; ++r)
        out[(size_t)(b0 + bb + r) * 49 + t] = acc[r];
    }
  }
}

// ---------------- weight pack kernel: f32 [i][o] -> bf16 [o][k], 1/sqrt(K) folded ----------------
__global__ void pack_weights_kernel(const float* __restrict__ w1, const float* __restrict__ w2,
                                    const float* __restrict__ w3, const float* __restrict__ w4,
                                    unsigned short* __restrict__ wt) {
  int i = blockIdx.x * blockDim.x + threadIdx.x;
  if (i >= 177152) return;
  const int NL[7]   = {320,64,64,32,32,32,32};
  const int KL1[7]  = {256,128,128,64,64,64,64};
  const int OFF1[8] = {0,81920,90112,98304,100352,102400,104448,106496};
  const int KL2[7]  = {64,64,64,32,32,32,32};
  const int OFF2[8] = {0,20480,24576,28672,29696,30720,31744,32768};
  if (i < 106496) {                                   // W1T
    int l = 0; while (i >= OFF1[l + 1]) ++l;
    int r = i - OFF1[l], K = KL1[l], o = r / K, k = r - o * K;
    float src = (l == 0) ? ((o < 64) ? w1[k * 64 + o] : w1[16384 + k * 256 + (o - 64)])
                         : w1[OFF1[l] + k * NL[l] + o];
    wt[i] = f2bf(src * rsqrtf((float)K));
  } else if (i < 172032) {                            // W2T then W3T (32768 halfs each)
    int j = i - 106496;
    const float* w = (j < 32768) ? w2 : w3;
    int jj = j & 32767;
    int l = 0; while (jj >= OFF2[l + 1]) ++l;
    int r = jj - OFF2[l], K = KL2[l], o = r / K, k = r - o * K;
    float src = (l == 0) ? ((o < 64) ? w[k * 64 + o] : w[4096 + k * 256 + (o - 64)])
                         : w[OFF2[l] + k * NL[l] + o];
    wt[i] = f2bf(src * rsqrtf((float)K));
  } else {                                            // W4 replicated: [l][16 cols][K]
    int j = i - 172032;                               // 5120 halfs
    const int OFF4P[8]   = {0,1024,2048,3072,3584,4096,4608,5120};
    const int W4SRC[7]   = {0,64,128,192,224,256,288};
    int l = 0; while (j >= OFF4P[l + 1]) ++l;
    int r = j - OFF4P[l], K = KL2[l];
    int k = r & (K - 1);                              // K is a power of two (64/32)
    wt[i] = f2bf(w4[W4SRC[l] + k] * rsqrtf((float)K));
  }
}

extern "C" void kernel_launch(void* const* d_in, const int* in_sizes, int n_in,
                              void* d_out, int out_size, void* d_ws, size_t ws_size,
                              hipStream_t stream) {
  const float* vraw = (const float*)d_in[0];
  const float* w1   = (const float*)d_in[1];
  const float* w2   = (const float*)d_in[2];
  const float* w3   = (const float*)d_in[3];
  const float* w4   = (const float*)d_in[4];
  unsigned short* wt = (unsigned short*)d_ws;         // 354,304 B of packed bf16 weights
  float* out = (float*)d_out;

  pack_weights_kernel<<<(177152 + 255) / 256, 256, 0, stream>>>(w1, w2, w3, w4, wt);

  const size_t SMEM = (67712 + 35968) * sizeof(unsigned short) + 4096 * sizeof(float); // 223,744 B
  (void)hipFuncSetAttribute(reinterpret_cast<const void*>(e3_fused_kernel),
                            hipFuncAttributeMaxDynamicSharedMemorySize, (int)SMEM);
  e3_fused_kernel<<<16384 / TILE_B, BLOCK, SMEM, stream>>>(vraw, wt, out);
}
